// MultiheadSelfAttention_33492154974510
// MI455X (gfx1250) — matrix operand order repaired and checked by reading, not compiled, not run on hardware
//
#include <hip/hip_runtime.h>
#include <hip/hip_bf16.h>

typedef __attribute__((ext_vector_type(16))) _Float16 v16h;
typedef __attribute__((ext_vector_type(8)))  _Float16 v8h;
typedef __attribute__((ext_vector_type(8)))  float    v8f;

#define EMBED 1024
#define SEQ   2048
#define NH    16
#define HD    64
#define ROWS  4096   // B*S

// ---------- WMMA fragment loaders (layouts per CDNA5 ISA 7.12.2) ----------

// A-matrix 16x32 f16, source row-major: element (m,k) at p[m*ld + k].
__device__ __forceinline__ v16h load_a16x32(const _Float16* p, int ld, int lane) {
    int r = lane & 15, hh = lane >> 4;
    v8h c0 = *(const v8h*)(p + r * ld + hh * 8);
    v8h c1 = *(const v8h*)(p + r * ld + 16 + hh * 8);
    v16h out;
#pragma unroll
    for (int i = 0; i < 8; i++) { out[i] = c0[i]; out[i + 8] = c1[i]; }
    return out;
}

// B-matrix 32x16 f16, source K-major: element (k,n) at p[n*ld + k].
__device__ __forceinline__ v16h load_b32x16(const _Float16* p, int ld, int lane) {
    int n = lane & 15, kg = lane >> 4;
    v8h c0 = *(const v8h*)(p + n * ld + kg * 8);
    v8h c1 = *(const v8h*)(p + n * ld + 16 + kg * 8);
    v16h out;
#pragma unroll
    for (int i = 0; i < 8; i++) { out[i] = c0[i]; out[i + 8] = c1[i]; }
    return out;
}

__device__ __forceinline__ v8f wmma_f16(v16h a, v16h b, v8f c) {
    return __builtin_amdgcn_wmma_f32_16x16x32_f16(false, a, false, b, (short)0, c, false, false);
}

// ---------- precision casts ----------

__global__ void cast_kernel(const float* __restrict__ src, _Float16* __restrict__ dst, int n) {
    int i = blockIdx.x * blockDim.x + threadIdx.x;
    if (i < n) dst[i] = (_Float16)src[i];
}

// W[k][n] (1024x1024) -> Wt[n][k] f16 so B-fragments are K-contiguous.
__global__ void transpose_cast_kernel(const float* __restrict__ W, _Float16* __restrict__ Wt) {
    int i = blockIdx.x * blockDim.x + threadIdx.x;   // i = k*1024 + n
    int k = i >> 10, n = i & 1023;
    Wt[n * 1024 + k] = (_Float16)W[i];
}

// ---------- fused QKV projection ----------
// grid (ROWS/32, 2, 3), block 256 (8 waves).
// Wave computes a 32Mx64N tile: 2 A-frags share 4 B-frags -> 8 WMMAs per
// 12 b128 loads per k-step; loads grouped ahead of the WMMA burst.
__global__ __launch_bounds__(256) void qkv_gemm_kernel(
                                const _Float16* __restrict__ Xh,
                                const _Float16* __restrict__ Wqt,
                                const _Float16* __restrict__ Wkt,
                                const _Float16* __restrict__ Wvt,
                                const float* __restrict__ bq,
                                const float* __restrict__ bk,
                                const float* __restrict__ bv,
                                _Float16* __restrict__ Q,
                                _Float16* __restrict__ Kc,
                                _Float16* __restrict__ Vt) {
    int lane = threadIdx.x & 31;
    int wave = threadIdx.x >> 5;
    int row0 = blockIdx.x * 32;
    int n0   = (blockIdx.y * 8 + wave) * 64;
    int which = blockIdx.z;
    const _Float16* Wt  = (which == 0) ? Wqt : (which == 1) ? Wkt : Wvt;
    const float*   bias = (which == 0) ? bq : (which == 1) ? bk : bv;

    const _Float16* Xp = Xh + (size_t)row0 * EMBED;

    v8f acc[2][4];
#pragma unroll
    for (int mt = 0; mt < 2; mt++)
#pragma unroll
        for (int t = 0; t < 4; t++)
#pragma unroll
            for (int i = 0; i < 8; i++) acc[mt][t][i] = 0.f;

    for (int kk = 0; kk < EMBED; kk += 32) {
        v16h a0 = load_a16x32(Xp + kk, EMBED, lane);
        v16h a1 = load_a16x32(Xp + (size_t)16 * EMBED + kk, EMBED, lane);
        v16h b[4];
#pragma unroll
        for (int t = 0; t < 4; t++)
            b[t] = load_b32x16(Wt + (size_t)(n0 + t * 16) * EMBED + kk, EMBED, lane);
#pragma unroll
        for (int t = 0; t < 4; t++) {
            acc[0][t] = wmma_f16(a0, b[t], acc[0][t]);
            acc[1][t] = wmma_f16(a1, b[t], acc[1][t]);
        }
    }

    int hh = lane >> 4, l16 = lane & 15;
#pragma unroll
    for (int t = 0; t < 4; t++) {
        int n = n0 + t * 16 + l16;
        int h = n >> 6, d = n & 63;
        float bb = bias[n];
#pragma unroll
        for (int mt = 0; mt < 2; mt++) {
#pragma unroll
            for (int r = 0; r < 8; r++) {
                int row = row0 + mt * 16 + r + 8 * hh;
                int b2 = row >> 11, s = row & 2047;
                float val = acc[mt][t][r] + bb;
                if (which == 0)
                    Q[(((size_t)(b2 * NH + h) * SEQ) + s) * HD + d] = (_Float16)val;
                else if (which == 1)
                    Kc[(((size_t)(b2 * NH + h) * SEQ) + s) * HD + d] = (_Float16)val;
                else  // V stored transposed [B,H,Hd,S] for K-contiguous P@V fragments
                    Vt[(((size_t)(b2 * NH + h) * HD) + d) * SEQ + s] = (_Float16)val;
            }
        }
    }
}

// ---------- causal flash attention, one wave per 16-query tile ----------
// grid 1024, block 128. Softmax runs in base-2 domain (v_exp_f32 is base-2).
__global__ __launch_bounds__(128) void attn_kernel(
                            const _Float16* __restrict__ Q,
                            const _Float16* __restrict__ Kc,
                            const _Float16* __restrict__ Vt,
                            _Float16* __restrict__ Oh) {
    __shared__ __align__(32) _Float16 ldsP[4][16][32];
    const float SCALE2 = 0.125f * 1.44269504089f;   // 1/sqrt(Hd) * log2(e)

    int lane = threadIdx.x & 31, wave = threadIdx.x >> 5;
    int tile = blockIdx.x * 4 + wave;   // 0..4095
    int bh = tile >> 7;                 // batch*head
    int q0 = (tile & 127) * 16;
    int b = bh >> 4, h = bh & 15;
    const _Float16* Qh = Q  + (size_t)bh * SEQ * HD;
    const _Float16* Kh = Kc + (size_t)bh * SEQ * HD;
    const _Float16* Vh = Vt + (size_t)bh * HD * SEQ;

    int hh = lane >> 4, l16 = lane & 15;
    v16h aQ0 = load_a16x32(Qh + q0 * HD + 0,  HD, lane);
    v16h aQ1 = load_a16x32(Qh + q0 * HD + 32, HD, lane);

    float m[8], l[8];
    v8f o[4];
#pragma unroll
    for (int r = 0; r < 8; r++) { m[r] = -1e30f; l[r] = 0.f; }
#pragma unroll
    for (int t = 0; t < 4; t++)
#pragma unroll
        for (int r = 0; r < 8; r++) o[t][r] = 0.f;

    v8f zero;
#pragma unroll
    for (int i = 0; i < 8; i++) zero[i] = 0.f;

    int nkb = (q0 + 16 + 31) >> 5;        // key blocks of 32, causal bound

    for (int kb = 0; kb < nkb; kb++) {
        int k0 = kb * 32;

        // K fragments + score WMMAs
        v16h bK0 = load_b32x16(Kh + (size_t)(k0 + 0)  * HD + 0,  HD, lane);
        v16h bK1 = load_b32x16(Kh + (size_t)(k0 + 0)  * HD + 32, HD, lane);
        v16h bK2 = load_b32x16(Kh + (size_t)(k0 + 16) * HD + 0,  HD, lane);
        v16h bK3 = load_b32x16(Kh + (size_t)(k0 + 16) * HD + 32, HD, lane);

        v8f s0 = zero, s1 = zero;
        s0 = wmma_f16(aQ0, bK0, s0);
        s0 = wmma_f16(aQ1, bK1, s0);
        s1 = wmma_f16(aQ0, bK2, s1);
        s1 = wmma_f16(aQ1, bK3, s1);

        // V fragments issued now; the softmax VALU section below hides latency
        v16h bV[4];
#pragma unroll
        for (int t = 0; t < 4; t++)
            bV[t] = load_b32x16(Vh + (size_t)(t * 16) * SEQ + k0, SEQ, lane);

        // scale (base-2) + causal mask (C-layout: lane col = l16, vgpr r -> row r+8*hh)
#pragma unroll
        for (int r = 0; r < 8; r++) {
            int row = q0 + r + 8 * hh;
            float v0 = s0[r] * SCALE2;
            float v1 = s1[r] * SCALE2;
            if (k0 + l16 > row)      v0 = -1e30f;
            if (k0 + 16 + l16 > row) v1 = -1e30f;
            s0[r] = v0; s1[r] = v1;
        }

        // online softmax in base-2, row reductions across 16 lanes of each half-wave
#pragma unroll
        for (int r = 0; r < 8; r++) {
            float mx = fmaxf(s0[r], s1[r]);
#pragma unroll
            for (int d = 1; d < 16; d <<= 1) mx = fmaxf(mx, __shfl_xor(mx, d, 16));
            float newm = fmaxf(m[r], mx);
            float alpha = exp2f(m[r] - newm);
            float p0 = exp2f(s0[r] - newm);
            float p1 = exp2f(s1[r] - newm);
            float sum = p0 + p1;
#pragma unroll
            for (int d = 1; d < 16; d <<= 1) sum += __shfl_xor(sum, d, 16);
            l[r] = l[r] * alpha + sum;
            m[r] = newm;
#pragma unroll
            for (int t = 0; t < 4; t++) o[t][r] *= alpha;
            int M = r + 8 * hh;
            ldsP[wave][M][l16]      = (_Float16)p0;
            ldsP[wave][M][16 + l16] = (_Float16)p1;
        }
        __builtin_amdgcn_wave_barrier();
        asm volatile("s_wait_dscnt 0x0" ::: "memory");
        // repack P (C-layout) -> A-layout via LDS, then P@V
        v16h aP = load_a16x32(&ldsP[wave][0][0], 32, lane);
        __builtin_amdgcn_wave_barrier();
#pragma unroll
        for (int t = 0; t < 4; t++)
            o[t] = wmma_f16(aP, bV[t], o[t]);
    }

    // normalize and store as f16 rows of [B*S, EMBED]
#pragma unroll
    for (int r = 0; r < 8; r++) {
        float inv = 1.0f / l[r];
        int rowg = b * SEQ + q0 + r + 8 * hh;
#pragma unroll
        for (int t = 0; t < 4; t++)
            Oh[(size_t)rowg * EMBED + h * HD + t * 16 + l16] = (_Float16)(o[t][r] * inv);
    }
}

// ---------- output projection, f32 result ----------
// grid (ROWS/32, 2), block 256 (8 waves), same 32x64 tiling as qkv.
__global__ __launch_bounds__(256) void oproj_gemm_kernel(
                                  const _Float16* __restrict__ Oh,
                                  const _Float16* __restrict__ Wot,
                                  const float* __restrict__ bo,
                                  float* __restrict__ out) {
    int lane = threadIdx.x & 31;
    int wave = threadIdx.x >> 5;
    int row0 = blockIdx.x * 32;
    int n0   = (blockIdx.y * 8 + wave) * 64;
    const _Float16* Xp = Oh + (size_t)row0 * EMBED;

    v8f acc[2][4];
#pragma unroll
    for (int mt = 0; mt < 2; mt++)
#pragma unroll
        for (int t = 0; t < 4; t++)
#pragma unroll
            for (int i = 0; i < 8; i++) acc[mt][t][i] = 0.f;

    for (int kk = 0; kk < EMBED; kk += 32) {
        v16h a0 = load_a16x32(Xp + kk, EMBED, lane);
        v16h a1 = load_a16x32(Xp + (size_t)16 * EMBED + kk, EMBED, lane);
        v16h b[4];
#pragma unroll
        for (int t = 0; t < 4; t++)
            b[t] = load_b32x16(Wot + (size_t)(n0 + t * 16) * EMBED + kk, EMBED, lane);
#pragma unroll
        for (int t = 0; t < 4; t++) {
            acc[0][t] = wmma_f16(a0, b[t], acc[0][t]);
            acc[1][t] = wmma_f16(a1, b[t], acc[1][t]);
        }
    }

    int hh = lane >> 4, l16 = lane & 15;
#pragma unroll
    for (int t = 0; t < 4; t++) {
        int n = n0 + t * 16 + l16;
        float bb = bo[n];
#pragma unroll
        for (int mt = 0; mt < 2; mt++) {
#pragma unroll
            for (int r = 0; r < 8; r++) {
                int row = row0 + mt * 16 + r + 8 * hh;
                out[(size_t)row * EMBED + n] = acc[mt][t][r] + bb;
            }
        }
    }
}

extern "C" void kernel_launch(void* const* d_in, const int* in_sizes, int n_in,
                              void* d_out, int out_size, void* d_ws, size_t ws_size,
                              hipStream_t stream) {
    const float* x  = (const float*)d_in[0];
    const float* Wq = (const float*)d_in[1];
    const float* bq = (const float*)d_in[2];
    const float* Wk = (const float*)d_in[3];
    const float* bk = (const float*)d_in[4];
    const float* Wv = (const float*)d_in[5];
    const float* bv = (const float*)d_in[6];
    const float* Wo = (const float*)d_in[7];
    const float* bo = (const float*)d_in[8];
    float* out = (float*)d_out;

    char* ws = (char*)d_ws;
    _Float16* Xh  = (_Float16*)ws; ws += (size_t)ROWS * EMBED * 2;
    _Float16* Wqt = (_Float16*)ws; ws += (size_t)EMBED * EMBED * 2;
    _Float16* Wkt = (_Float16*)ws; ws += (size_t)EMBED * EMBED * 2;
    _Float16* Wvt = (_Float16*)ws; ws += (size_t)EMBED * EMBED * 2;
    _Float16* Wot = (_Float16*)ws; ws += (size_t)EMBED * EMBED * 2;
    _Float16* Qb  = (_Float16*)ws; ws += (size_t)ROWS * EMBED * 2;
    _Float16* Kb  = (_Float16*)ws; ws += (size_t)ROWS * EMBED * 2;
    _Float16* Vtb = (_Float16*)ws; ws += (size_t)ROWS * EMBED * 2;
    _Float16* Ohb = (_Float16*)ws; ws += (size_t)ROWS * EMBED * 2;

    cast_kernel<<<(ROWS * EMBED) / 256, 256, 0, stream>>>(x, Xh, ROWS * EMBED);
    transpose_cast_kernel<<<(EMBED * EMBED) / 256, 256, 0, stream>>>(Wq, Wqt);
    transpose_cast_kernel<<<(EMBED * EMBED) / 256, 256, 0, stream>>>(Wk, Wkt);
    transpose_cast_kernel<<<(EMBED * EMBED) / 256, 256, 0, stream>>>(Wv, Wvt);
    transpose_cast_kernel<<<(EMBED * EMBED) / 256, 256, 0, stream>>>(Wo, Wot);

    qkv_gemm_kernel<<<dim3(ROWS / 32, 2, 3), 256, 0, stream>>>(
        Xh, Wqt, Wkt, Wvt, bq, bk, bv, Qb, Kb, Vtb);

    attn_kernel<<<1024, 128, 0, stream>>>(Qb, Kb, Vtb, Ohb);

    oproj_gemm_kernel<<<dim3(ROWS / 32, 2), 256, 0, stream>>>(Ohb, Wot, bo, out);
}
